// ModulateConvLayer_79817672229337
// MI455X (gfx1250) — compile-verified, matrix operand order checked by reading, not run on hardware
//
#include <hip/hip_runtime.h>
#include <math.h>

// ---------------------------------------------------------------------------
// Types for CDNA5 WMMA (wave32): v_wmma_f32_16x16x32_bf16
// ---------------------------------------------------------------------------
typedef __attribute__((ext_vector_type(16))) __bf16 v16bf;
typedef __attribute__((ext_vector_type(8)))  __bf16 v8bf;
typedef __attribute__((ext_vector_type(8)))  float  v8f;

union V16 { v16bf v; v8bf h[2]; };

#define WSCALE        (1.0f / 48.0f)            /* sqrt(2)/sqrt(9*512) exactly */
#define STYLE_WSCALE  0.04419417382415922f      /* 1/sqrt(512) */
#define EPS           1e-8f

#define B_DIM   16
#define C_DIM   512
#define M_TOT   65536     /* B*H*W */
#define OUT_X   33554432  /* 16*512*64*64 */

#define WS_XU    ((size_t)0)
#define WS_W2    ((size_t)67108864)              /* 64 MB  */
#define WS_CONV  ((size_t)71827456)              /* +4.5MB */
#define WS_ZBUF  ((size_t)138936320)             /* +64MB  */
#define ZBUF_N   1088                            /* bf16 zeros: covers K-range+frag */

// ---------------------------------------------------------------------------
// 1) style = w @ (style_weight * sw).T + style_bias   -> d_out tail [16,1024]
// ---------------------------------------------------------------------------
__global__ __launch_bounds__(256) void style_k(const float* __restrict__ w,
                                               const float* __restrict__ sw,
                                               const float* __restrict__ sb,
                                               float* __restrict__ style) {
  int gid = blockIdx.x * 256 + threadIdx.x;     // 0..16383
  int b = gid >> 10, j = gid & 1023;
  const float* wr = w  + b * 512;
  const float* sr = sw + (size_t)j * 512;
  float acc = 0.f;
  #pragma unroll 8
  for (int k = 0; k < 512; ++k) acc += wr[k] * sr[k];
  style[gid] = acc * STYLE_WSCALE + sb[j];
}

// ---------------------------------------------------------------------------
// 2) NCHW f32 [16,512,32,32] -> upsampled NHWC bf16 [16,64,64,512]
// ---------------------------------------------------------------------------
__global__ __launch_bounds__(256) void upsample_k(const float* __restrict__ x,
                                                  __bf16* __restrict__ xu) {
  int pix = blockIdx.x;                 // b*4096 + h*64 + w
  int b = pix >> 12, hw = pix & 4095;
  int h = hw >> 6,   ww = hw & 63;
  const float* src = x + (size_t)b * 512 * 1024 + (h >> 1) * 32 + (ww >> 1);
  __bf16* dst = xu + ((size_t)pix << 9);
  for (int c = threadIdx.x; c < 512; c += 256)
    dst[c] = (__bf16)src[(size_t)c * 1024];
}

// ---------------------------------------------------------------------------
// 3) weight[co][ci][3][3]*WSCALE -> bf16 w2[tap][co][ci]; also zero the guard
// ---------------------------------------------------------------------------
__global__ __launch_bounds__(256) void wprep_k(const float* __restrict__ wt,
                                               __bf16* __restrict__ w2,
                                               __bf16* __restrict__ zbuf) {
  if (blockIdx.x == 0) {                        // zero-guard for OOB A rows
    for (int i = threadIdx.x; i < ZBUF_N; i += 256) zbuf[i] = (__bf16)0.0f;
  }
  int idx = blockIdx.x * 256 + threadIdx.x;     // output-linear
  if (idx >= 9 * 512 * 512) return;
  int ci = idx & 511, rest = idx >> 9;
  int co = rest & 511, tap = rest >> 9;
  w2[idx] = (__bf16)(wt[((size_t)co * 512 + ci) * 9 + tap] * WSCALE);
}

// ---------------------------------------------------------------------------
// 4) Implicit-GEMM conv via v_wmma_f32_16x16x32_bf16.
//    M = b*4096+h*64+w (65536), N = co (512), K = tap-major (9 x 512 ci).
//    Block: 256 thr = 8 waves (2 M x 4 N). Wave tile: 64(M) x 64(N) = 4x4 accs
//    -> 16 wmma per 16 fragment loads per K-step of 32.
//    Zero-pad handled via per-lane pointer select to a zeroed guard buffer
//    (no EXEC divergence, no zero-splats in the hot loop).
// ---------------------------------------------------------------------------
__global__ __launch_bounds__(256) void conv_gemm_k(const __bf16* __restrict__ xu,
                                                   const __bf16* __restrict__ w2,
                                                   const __bf16* __restrict__ zbuf,
                                                   __bf16* __restrict__ conv) {
  const int lane  = threadIdx.x & 31;
  const int wave  = threadIdx.x >> 5;
  const int wm    = wave >> 2;                  // 0..1
  const int wn    = wave & 3;                   // 0..3
  const int m0    = blockIdx.x * 128 + wm * 64; // 4 M sub-tiles of 16
  const int n0    = blockIdx.y * 256 + wn * 64; // 4 N sub-tiles of 16
  const int lhalf = lane >> 4;                  // K-chunk select (ISA A/B layout)
  const int lmod  = lane & 15;

  // Per-lane M rows of the 4 A sub-tiles (fixed for whole kernel)
  int bb[4], hh[4], ww[4];
  #pragma unroll
  for (int mi = 0; mi < 4; ++mi) {
    int m = m0 + mi * 16 + lmod;
    bb[mi] = m >> 12;
    int hw = m & 4095;
    hh[mi] = hw >> 6;
    ww[mi] = hw & 63;
  }

  v8f acc[4][4] = {};

  for (int tap = 0; tap < 9; ++tap) {
    const int kh = tap / 3 - 1, kw = tap % 3 - 1;

    // Per-tap A row pointers; OOB rows redirected to zero-guard (v_cndmask,
    // once per tap — inner loop loads are unconditional).
    const __bf16* ap[4];
    #pragma unroll
    for (int mi = 0; mi < 4; ++mi) {
      int ih = hh[mi] + kh, iw = ww[mi] + kw;
      bool valid = ((unsigned)ih < 64u) && ((unsigned)iw < 64u);
      const __bf16* real =
          xu + (((size_t)(bb[mi] << 12) + ih * 64 + iw) << 9) + lhalf * 8;
      ap[mi] = valid ? real : zbuf;
    }
    // B row pointers: w2[tap][co][ci]
    const __bf16* bp[4];
    #pragma unroll
    for (int ni = 0; ni < 4; ++ni) {
      int co = n0 + ni * 16 + lmod;
      bp[ni] = w2 + (((size_t)tap * 512 + co) << 9) + lhalf * 8;
    }

    #pragma unroll 4
    for (int ci = 0; ci < 512; ci += 32) {
      V16 a4[4], b4[4];
      #pragma unroll
      for (int mi = 0; mi < 4; ++mi) {
        a4[mi].h[0] = *(const v8bf*)(ap[mi] + ci);
        a4[mi].h[1] = *(const v8bf*)(ap[mi] + ci + 16);
      }
      #pragma unroll
      for (int ni = 0; ni < 4; ++ni) {
        b4[ni].h[0] = *(const v8bf*)(bp[ni] + ci);
        b4[ni].h[1] = *(const v8bf*)(bp[ni] + ci + 16);
      }
      #pragma unroll
      for (int mi = 0; mi < 4; ++mi) {
        #pragma unroll
        for (int ni = 0; ni < 4; ++ni) {
          acc[mi][ni] = __builtin_amdgcn_wmma_f32_16x16x32_bf16(
              /*neg_a=*/false, a4[mi].v, /*neg_b=*/false, b4[ni].v,
              /*c_mod=*/(short)0, acc[mi][ni],
              /*reuse_a=*/false, /*reuse_b=*/false);
        }
      }
    }
  }

  // Epilogue: D layout lane l -> N = l%16, rows M = r + 8*(l/16); store NCHW bf16
  #pragma unroll
  for (int mi = 0; mi < 4; ++mi) {
    #pragma unroll
    for (int ni = 0; ni < 4; ++ni) {
      int n = n0 + ni * 16 + lmod;
      #pragma unroll
      for (int r = 0; r < 8; ++r) {
        int m = m0 + mi * 16 + lhalf * 8 + r;
        int b = m >> 12, hw = m & 4095;
        conv[(((size_t)b * 512 + n) << 12) + hw] = (__bf16)acc[mi][ni][r];
      }
    }
  }
}

// ---------------------------------------------------------------------------
// 5) blur(3x3, [1,2,1]^2/16, zero-pad) + noise + bias + lrelu + instance-norm
//    + style affine; one workgroup per (b, co) plane.
// ---------------------------------------------------------------------------
__global__ __launch_bounds__(256) void post_k(const __bf16* __restrict__ conv,
                                              const float* __restrict__ noise,
                                              const float* __restrict__ nstr,
                                              const float* __restrict__ bias,
                                              const float* __restrict__ style,
                                              float* __restrict__ out) {
  const int co = blockIdx.x, b = blockIdx.y, tid = threadIdx.x;
  __shared__ float plane[4096];
  __shared__ float sred[256], sred2[256];
  __shared__ float smean, srstd;

  const __bf16* src = conv + (((size_t)b * 512 + co) << 12);
  for (int i = tid; i < 4096; i += 256) plane[i] = (float)src[i];
  __syncthreads();

  const float ns = nstr[co], bs = bias[co];
  const float BW[3][3] = {{1.f/16, 2.f/16, 1.f/16},
                          {2.f/16, 4.f/16, 2.f/16},
                          {1.f/16, 2.f/16, 1.f/16}};
  float vals[16];
  float s = 0.f, s2 = 0.f;
  #pragma unroll
  for (int e = 0; e < 16; ++e) {
    int px = e * 256 + tid;
    int h = px >> 6, w = px & 63;
    float a = 0.f;
    #pragma unroll
    for (int dr = -1; dr <= 1; ++dr) {
      #pragma unroll
      for (int dc = -1; dc <= 1; ++dc) {
        int ih = h + dr, iw = w + dc;
        if ((unsigned)ih < 64u && (unsigned)iw < 64u)
          a += BW[dr + 1][dc + 1] * plane[ih * 64 + iw];
      }
    }
    float v = a + noise[px] * ns + bs;
    v = v > 0.f ? v : 0.2f * v;          // leaky relu
    vals[e] = v;
    s += v; s2 += v * v;
  }

  sred[tid] = s; sred2[tid] = s2;
  __syncthreads();
  for (int off = 128; off > 0; off >>= 1) {
    if (tid < off) { sred[tid] += sred[tid + off]; sred2[tid] += sred2[tid + off]; }
    __syncthreads();
  }
  if (tid == 0) {
    float mean = sred[0] * (1.f / 4096.f);
    float var  = sred2[0] * (1.f / 4096.f) - mean * mean;
    smean = mean;
    srstd = rsqrtf(var + EPS);
  }
  __syncthreads();

  const float mean = smean, rstd = srstd;
  const float sc = style[(size_t)b * 1024 + co] + 1.0f;
  const float sh = style[(size_t)b * 1024 + 512 + co];
  float* dst = out + (((size_t)b * 512 + co) << 12);
  #pragma unroll
  for (int e = 0; e < 16; ++e) {
    int px = e * 256 + tid;
    dst[px] = (vals[e] - mean) * rstd * sc + sh;
  }
}

// ---------------------------------------------------------------------------
// launch
// ---------------------------------------------------------------------------
extern "C" void kernel_launch(void* const* d_in, const int* in_sizes, int n_in,
                              void* d_out, int out_size, void* d_ws, size_t ws_size,
                              hipStream_t stream) {
  const float* x      = (const float*)d_in[0];  // [16,512,32,32]
  const float* w      = (const float*)d_in[1];  // [16,512]
  const float* weight = (const float*)d_in[2];  // [512,512,3,3]
  const float* bias   = (const float*)d_in[3];  // [512]
  const float* swgt   = (const float*)d_in[4];  // [1024,512]
  const float* sbias  = (const float*)d_in[5];  // [1024]
  const float* noise  = (const float*)d_in[6];  // [1,1,64,64]
  const float* nstr   = (const float*)d_in[7];  // [1,512,1,1]

  float* out   = (float*)d_out;
  float* style = out + OUT_X;                   // d_out tail: [16,1024]

  char* ws = (char*)d_ws;
  __bf16* xu   = (__bf16*)(ws + WS_XU);         // 64 MB NHWC bf16 (upsampled)
  __bf16* w2   = (__bf16*)(ws + WS_W2);         // 4.5 MB [tap][co][ci]
  __bf16* conv = (__bf16*)(ws + WS_CONV);       // 64 MB NCHW bf16
  __bf16* zbuf = (__bf16*)(ws + WS_ZBUF);       // 2.1 KB zero guard

  // 1) style GEMM -> d_out tail
  style_k<<<64, 256, 0, stream>>>(w, swgt, sbias, style);
  // 2) upsample + bf16 + NHWC
  upsample_k<<<M_TOT, 256, 0, stream>>>(x, xu);
  // 3) weight scale + bf16 + [tap][co][ci]; zero guard init
  wprep_k<<<(9 * 512 * 512) / 256, 256, 0, stream>>>(weight, w2, zbuf);
  // 4) WMMA implicit-GEMM conv: grid = (M/128, N/256)
  conv_gemm_k<<<dim3(M_TOT / 128, C_DIM / 256), 256, 0, stream>>>(xu, w2, zbuf, conv);
  // 5) blur + noise + bias + lrelu + instnorm + style -> d_out
  post_k<<<dim3(C_DIM, B_DIM), 256, 0, stream>>>(conv, noise, nstr, bias, style, out);
}
